// RNN_34505767256242
// MI455X (gfx1250) — compile-verified, matrix operand order checked by reading, not run on hardware
//
#include <hip/hip_runtime.h>
#include <hip/hip_bf16.h>
#include <math.h>

// ---------------------------------------------------------------------------
// LSTM LM on MI455X (gfx1250, wave32, WMMA 16x16x32 f16 -> f32 accum)
//   VOCAB=32000 EMB=128 HID=256 B=16 S=512
// Stage 1: xg[s*16+b, 0:1024] = emb[x[b,s]] @ w_ih^T + (b_ih+b_hh)   (WMMA)
// Stage 2: single-WGP sequential LSTM scan, w_hh resident in VGPRs   (WMMA)
// Stage 3: logits = h @ w_out^T + b_out, 128x128 blocking, NT stores (WMMA)
// ---------------------------------------------------------------------------

typedef __attribute__((ext_vector_type(16))) _Float16 v16h;
typedef __attribute__((ext_vector_type(2)))  _Float16 v2h;
typedef __attribute__((ext_vector_type(8)))  float    v8f;
typedef __attribute__((ext_vector_type(4)))  int      v4i;

#define WMMA_F32_F16(A, B, C) \
  __builtin_amdgcn_wmma_f32_16x16x32_f16(false, (A), false, (B), (short)0, (C), false, false)

#if __has_builtin(__builtin_amdgcn_global_load_async_to_lds_b128) && \
    __has_builtin(__builtin_amdgcn_s_wait_asynccnt)
#define HAVE_ASYNC_LDS 1
typedef __attribute__((address_space(1))) v4i* as1_v4ip;  // global int4*
typedef __attribute__((address_space(3))) v4i* as3_v4ip;  // LDS int4*
#else
#define HAVE_ASYNC_LDS 0
#endif

// Load a 16x32 f16 fragment (A-layout; also used for B since all weights are
// stored [N,K] row-major, i.e. B columns are contiguous in K).
// ISA layout: lanes 0-15 -> row = lane, K = k0 + {0..7, 16..23}
//             lanes 16-31 -> row = lane-16, K = k0 + {8..15, 24..31}
// Written as dword pair loads; compiler merges into 2 x b128 per fragment.
__device__ __forceinline__ v16h frag16(const _Float16* __restrict__ base,
                                       int ld, int k0, int lane) {
  const int row = lane & 15;
  const int klo = k0 + ((lane & 16) ? 8 : 0);
  const _Float16* r = base + row * ld;
  v16h out;
#pragma unroll
  for (int p = 0; p < 8; ++p) {
    const int k = klo + 2 * p + ((p >= 4) ? 8 : 0);
    v2h h2 = *reinterpret_cast<const v2h*>(r + k);
    out[2 * p]     = h2.x;
    out[2 * p + 1] = h2.y;
  }
  return out;
}

__device__ __forceinline__ float sigf(float x) {
  return 1.0f / (1.0f + __expf(-x));
}

// ---------------------------------------------------------------- convert ---
__global__ void lm_cvt_f32_f16(const float* __restrict__ src,
                               _Float16* __restrict__ dst, int n) {
  for (int i = blockIdx.x * blockDim.x + threadIdx.x; i < n;
       i += gridDim.x * blockDim.x)
    dst[i] = (_Float16)src[i];
}

// ------------------------------------------------------- stage 1: xg GEMM ---
// grid = (512 /*s*/, 16 /*n-groups of 4 tiles*/), block = 128 (4 waves)
__global__ void lm_xg_gemm(const int* __restrict__ x,
                           const float* __restrict__ emb,
                           const _Float16* __restrict__ w_ih16,
                           const float* __restrict__ b_ih,
                           const float* __restrict__ b_hh,
                           float* __restrict__ xg) {
  __shared__ _Float16 At[16 * 128];  // rows = batch b, cols = EMB
  const int tid  = threadIdx.x;
  const int lane = tid & 31;
  const int wv   = tid >> 5;
  const int s    = blockIdx.x;

  // gather + convert 16 embedding rows (token per batch element at time s)
  {
    const int r  = tid >> 3;         // 0..15 : batch row
    const int c0 = (tid & 7) * 16;   // 16 f32 each
    const int tok = x[r * 512 + s];
    const float* e = emb + (size_t)tok * 128 + c0;
#pragma unroll
    for (int i = 0; i < 16; ++i) At[r * 128 + c0 + i] = (_Float16)e[i];
  }
  __syncthreads();

  const int n0  = (blockIdx.y * 4 + wv) * 16;
  const int col = lane & 15;
  const int hi8 = (lane & 16) ? 8 : 0;

  const float bias = b_ih[n0 + col] + b_hh[n0 + col];
  v8f acc;
#pragma unroll
  for (int r = 0; r < 8; ++r) acc[r] = bias;

  const _Float16* bb = w_ih16 + n0 * 128;  // B col n = w_ih row n (contig K)
#pragma unroll
  for (int kk = 0; kk < 4; ++kk) {
    v16h a = frag16(At, 128, kk * 32, lane);
    v16h b = frag16(bb, 128, kk * 32, lane);
    acc = WMMA_F32_F16(a, b, acc);
  }
#pragma unroll
  for (int r = 0; r < 8; ++r)
    xg[(size_t)(s * 16 + r + hi8) * 1024 + n0 + col] = acc[r];
}

// -------------------------------------------------- stage 2: LSTM scan ------
// Single workgroup: 1024 threads = 32 waves; wave w owns gate columns
// [w*32, w*32+32). w_hh fragments live in VGPRs for the whole scan.
__global__ void __launch_bounds__(1024)
lm_lstm_scan(const float* __restrict__ xg,
             const _Float16* __restrict__ w_hh16,
             _Float16* __restrict__ h_all) {
  __shared__ _Float16 hbuf[16 * 256];   // 8 KB
  __shared__ float    cbuf[16 * 256];   // 16 KB
  __shared__ _Float16 gbuf[16 * 1024];  // 32 KB gates staging

  const int tid  = threadIdx.x;
  const int lane = tid & 31;
  const int wv   = tid >> 5;
  const int n0   = wv * 32;
  const int col  = lane & 15;
  const int hi8  = (lane & 16) ? 8 : 0;

  // zero-init h, c
  for (int e = tid; e < 16 * 256; e += 1024) { hbuf[e] = (_Float16)0.0f; cbuf[e] = 0.0f; }

  // preload w_hh fragments: 2 N-tiles x 8 K-steps, 128 VGPRs of B per wave
  v16h Br0[8], Br1[8];
#pragma unroll
  for (int kk = 0; kk < 8; ++kk) {
    Br0[kk] = frag16(w_hh16 + (n0 +  0) * 256, 256, kk * 32, lane);
    Br1[kk] = frag16(w_hh16 + (n0 + 16) * 256, 256, kk * 32, lane);
  }
  __syncthreads();

  for (int s = 0; s < 512; ++s) {
    // C init = precomputed input gates (bias already folded in)
    const float* xr = xg + (size_t)s * 16 * 1024;
    v8f acc0, acc1;
#pragma unroll
    for (int r = 0; r < 8; ++r) {
      const int b = r + hi8;
      acc0[r] = xr[b * 1024 + n0 + col];
      acc1[r] = xr[b * 1024 + n0 + 16 + col];
    }
    // gates += h @ w_hh^T   (A = h from LDS, reused across both N-tiles)
#pragma unroll
    for (int kk = 0; kk < 8; ++kk) {
      v16h a = frag16(hbuf, 256, kk * 32, lane);
      acc0 = WMMA_F32_F16(a, Br0[kk], acc0);
      acc1 = WMMA_F32_F16(a, Br1[kk], acc1);
    }
    // stage gates to LDS (f16)
#pragma unroll
    for (int r = 0; r < 8; ++r) {
      const int b = r + hi8;
      gbuf[b * 1024 + n0 + col]      = (_Float16)acc0[r];
      gbuf[b * 1024 + n0 + 16 + col] = (_Float16)acc1[r];
    }
    __syncthreads();

    // elementwise: 4096 cell updates, 4 per thread
#pragma unroll
    for (int j = 0; j < 4; ++j) {
      const int e  = tid + j * 1024;
      const int b  = e >> 8;
      const int cc = e & 255;
      const float iv = (float)gbuf[b * 1024 +       cc];
      const float fv = (float)gbuf[b * 1024 + 256 + cc];
      const float gv = (float)gbuf[b * 1024 + 512 + cc];
      const float ov = (float)gbuf[b * 1024 + 768 + cc];
      float c = sigf(fv) * cbuf[e] + sigf(iv) * tanhf(gv);
      const float h = sigf(ov) * tanhf(c);
      cbuf[e] = c;
      hbuf[e] = (_Float16)h;
      h_all[(size_t)(b * 512 + s) * 256 + cc] = (_Float16)h;
    }
    __syncthreads();
  }
}

// ----------------------------------------------- stage 3: logits GEMM -------
// Workgroup = 128 rows x 128 cols of logits. 8 waves; wave wv owns N-tile
// n0 = (bx*8+wv)*16 and computes 8 M-tiles against it (8x B-fragment reuse).
// A tile (128x256 f16 = 64 KB) staged in LDS via async global->LDS loads.
// grid = (250 /*n-groups*/, 64 /*m-groups*/), block = 256 (8 waves)
__global__ void __launch_bounds__(256)
lm_out_gemm(const _Float16* __restrict__ h_all,
            const _Float16* __restrict__ w_out16,
            const float* __restrict__ b_out,
            float* __restrict__ out) {
  __shared__ _Float16 At[128 * 256];  // 64 KB
  const int tid  = threadIdx.x;
  const int lane = tid & 31;
  const int wv   = tid >> 5;
  const int m0   = blockIdx.y * 128;
  const int n0   = (blockIdx.x * 8 + wv) * 16;

  // stage A tile: 64 KB = 4096 x 16B chunks; 16 chunks per thread, coalesced
  {
    const _Float16* src = h_all + (size_t)m0 * 256;
#pragma unroll
    for (int i = 0; i < 16; ++i) {
      const int c = tid + i * 256;  // chunk index (16B each)
#if HAVE_ASYNC_LDS
      __builtin_amdgcn_global_load_async_to_lds_b128(
          (as1_v4ip)(void*)(src + c * 8), (as3_v4ip)(void*)(&At[c * 8]), 0, 0);
#else
      *reinterpret_cast<uint4*>(&At[c * 8]) =
          *reinterpret_cast<const uint4*>(src + c * 8);
#endif
    }
#if HAVE_ASYNC_LDS
    __builtin_amdgcn_s_wait_asynccnt(0);
#endif
  }
  __syncthreads();

  const int col = lane & 15;
  const int hi8 = (lane & 16) ? 8 : 0;
  const float bias = b_out[n0 + col];
  v8f acc[8];
#pragma unroll
  for (int mt = 0; mt < 8; ++mt)
#pragma unroll
    for (int r = 0; r < 8; ++r) acc[mt][r] = bias;

  const _Float16* bb = w_out16 + (size_t)n0 * 256;  // col n = w_out row n
#pragma unroll
  for (int kk = 0; kk < 8; ++kk) {
    __builtin_prefetch(bb + 256 * ((kk + 1) & 7), 0, 1);
    v16h b = frag16(bb, 256, kk * 32, lane);
#pragma unroll
    for (int mt = 0; mt < 8; ++mt) {
      v16h a = frag16(At + mt * 16 * 256, 256, kk * 32, lane);
      acc[mt] = WMMA_F32_F16(a, b, acc[mt]);
    }
  }

  // non-temporal stores: 1.05 GB of logits, never re-read -> don't pollute L2
#pragma unroll
  for (int mt = 0; mt < 8; ++mt) {
#pragma unroll
    for (int r = 0; r < 8; ++r) {
      float* p = out + (size_t)(m0 + mt * 16 + r + hi8) * 32000 + n0 + col;
#if __has_builtin(__builtin_nontemporal_store)
      __builtin_nontemporal_store(acc[mt][r], p);
#else
      *p = acc[mt][r];
#endif
    }
  }
}

// ---------------------------------------------------------------------------
extern "C" void kernel_launch(void* const* d_in, const int* in_sizes, int n_in,
                              void* d_out, int out_size, void* d_ws, size_t ws_size,
                              hipStream_t stream) {
  const int*   x     = (const int*)d_in[0];
  const float* emb   = (const float*)d_in[1];
  const float* w_ih  = (const float*)d_in[2];
  const float* w_hh  = (const float*)d_in[3];
  const float* b_ih  = (const float*)d_in[4];
  const float* b_hh  = (const float*)d_in[5];
  const float* w_out = (const float*)d_in[6];
  const float* b_out = (const float*)d_in[7];
  float* out = (float*)d_out;

  // workspace layout (~55 MB)
  char* ws = (char*)d_ws;
  _Float16* w_ih16  = (_Float16*)(ws);                 // 1024*128*2  = 256 KB
  _Float16* w_hh16  = (_Float16*)(ws + 262144);        // 1024*256*2  = 512 KB
  _Float16* w_out16 = (_Float16*)(ws + 786432);        // 32000*256*2 = 16.4 MB
  float*    xg      = (float*)   (ws + 17170432);      // 8192*1024*4 = 33.5 MB
  _Float16* h_all   = (_Float16*)(ws + 50724864);      // 8192*256*2  = 4 MB

  lm_cvt_f32_f16<<<256, 256, 0, stream>>>(w_ih, w_ih16, 1024 * 128);
  lm_cvt_f32_f16<<<512, 256, 0, stream>>>(w_hh, w_hh16, 1024 * 256);
  lm_cvt_f32_f16<<<2048, 256, 0, stream>>>(w_out, w_out16, 32000 * 256);

  lm_xg_gemm<<<dim3(512, 16), 128, 0, stream>>>(x, emb, w_ih16, b_ih, b_hh, xg);
  lm_lstm_scan<<<1, 1024, 0, stream>>>(xg, w_hh16, h_all);
  lm_out_gemm<<<dim3(250, 64), 256, 0, stream>>>(h_all, w_out16, b_out, out);
}
